// GeoSamplerModel_2654289789698
// MI455X (gfx1250) — compile-verified, matrix operand order checked by reading, not run on hardware
//
#include <hip/hip_runtime.h>
#include <hip/hip_bf16.h>

// ---------------- model constants ----------------
#define C_DIM   1024
#define C2      1026        // C + 2
#define C2P     1056        // diff K padded to 33*32
#define OPAD    1040        // diff out cols padded to 65*16
#define G2      2052        // 2*C2
#define G2P     2080        // agg K padded to 65*32
#define B_SZ    32
#define N0      512
#define N1      256
#define N2      64
#define KNN     16
#define OUT_DIM 3584
#define LN_EPS  1e-5f

typedef __attribute__((ext_vector_type(16))) __bf16 bf16x16;
typedef __attribute__((ext_vector_type(8)))  __bf16 bf16x8;
typedef __attribute__((ext_vector_type(8)))  float  f32x8;
typedef __attribute__((ext_vector_type(4)))  unsigned int u32x4;
typedef __attribute__((ext_vector_type(8)))  int i32x8;
typedef __attribute__((ext_vector_type(4)))  int i32x4;

__device__ __forceinline__ int imin(int a, int b) { return a < b ? a : b; }
__device__ __forceinline__ int imax(int a, int b) { return a > b ? a : b; }

// CDNA5 16-bit A 16x32 fragment layout: lanes<16 hold K {kb+0..7, kb+16..23},
// lanes>=16 hold K {kb+8..15, kb+24..31}  -> two contiguous 8x bf16 runs.
__device__ __forceinline__ bf16x16
frag_load(const __bf16* __restrict__ base, int row, int rowlen, int kbase, int lane) {
    const int half = (lane >> 4) & 1;
    const __bf16* p = base + (size_t)row * rowlen + kbase + half * 8;
    union U { bf16x16 v; bf16x8 h[2]; } u;
    u.h[0] = *(const bf16x8*)(p);
    u.h[1] = *(const bf16x8*)(p + 16);
    return u.v;
}

__device__ __forceinline__ f32x8
wmma_bf16(bf16x16 a, bf16x16 b, f32x8 c) {
    return __builtin_amdgcn_wmma_f32_16x16x32_bf16(false, a, false, b, (short)0, c,
                                                   false, false);
}

// ---------------- weight pad/convert to bf16 ----------------
__global__ void pad_bf16_kernel(const float* __restrict__ src, __bf16* __restrict__ dst,
                                int sr, int sc, int dr, int dc) {
    int i = blockIdx.x * blockDim.x + threadIdx.x;
    if (i >= dr * dc) return;
    int r = i / dc, c = i % dc;
    float v = (r < sr && c < sc) ? src[(size_t)r * sc + c] : 0.f;
    dst[i] = (__bf16)v;
}

// ---------------- bilinear sampling ----------------
// grid = 2*points[...,::-1]-1  ==>  x = p1*(W-1), y = p0*(H-1)
__global__ void bilinear_kernel(const float* __restrict__ points,
                                const float* __restrict__ feat,
                                float* __restrict__ fts0) {
    const int bn = blockIdx.x;           // b*N0 + n
    const float p0 = points[(size_t)bn * 2 + 0];
    const float p1 = points[(size_t)bn * 2 + 1];
    const float x = p1 * 31.f, y = p0 * 31.f;
    const float x0 = floorf(x), y0 = floorf(y);
    const float wx = x - x0, wy = y - y0;
    const int x0i = imin(imax((int)x0, 0), 31);
    const int x1i = imin(imax((int)x0 + 1, 0), 31);
    const int y0i = imin(imax((int)y0, 0), 31);
    const int y1i = imin(imax((int)y0 + 1, 0), 31);
    const float w00 = (1.f - wx) * (1.f - wy);
    const float w01 = wx * (1.f - wy);
    const float w10 = (1.f - wx) * wy;
    const float w11 = wx * wy;
    for (int c = threadIdx.x; c < C_DIM; c += 256) {
        const float* fc = feat + (size_t)c * 1024;
        float v = fc[y0i * 32 + x0i] * w00 + fc[y0i * 32 + x1i] * w01 +
                  fc[y1i * 32 + x0i] * w10 + fc[y1i * 32 + x1i] * w11;
        fts0[(size_t)bn * C_DIM + c] = v;
    }
}

// ---------------- farthest point sampling (serial scan, 1 block / batch) ----
__global__ void fps_kernel(const float* __restrict__ pts, int NP, int NS,
                           int* __restrict__ fps_idx, float* __restrict__ pts_out) {
    __shared__ float ptsx[512], ptsy[512], dist[512], rv[512];
    __shared__ int ri[512];
    __shared__ int farL;
    const int tid = threadIdx.x;         // blockDim == NP (power of 2)
    const int b = blockIdx.x;
    ptsx[tid] = pts[((size_t)b * NP + tid) * 2 + 0];
    ptsy[tid] = pts[((size_t)b * NP + tid) * 2 + 1];
    dist[tid] = 1e10f;
    if (tid == 0) farL = 0;
    __syncthreads();
    for (int t = 0; t < NS; ++t) {
        const int far = farL;
        if (tid == 0) {
            fps_idx[(size_t)b * NS + t] = far;
            pts_out[((size_t)b * NS + t) * 2 + 0] = ptsx[far];
            pts_out[((size_t)b * NS + t) * 2 + 1] = ptsy[far];
        }
        const float dx = ptsx[tid] - ptsx[far];
        const float dy = ptsy[tid] - ptsy[far];
        dist[tid] = fminf(dist[tid], dx * dx + dy * dy);
        rv[tid] = dist[tid]; ri[tid] = tid;
        __syncthreads();
        for (int s2 = NP >> 1; s2 > 0; s2 >>= 1) {
            if (tid < s2 && rv[tid + s2] > rv[tid]) { rv[tid] = rv[tid + s2]; ri[tid] = ri[tid + s2]; }
            __syncthreads();
        }
        if (tid == 0) farL = ri[0];
        __syncthreads();
    }
}

// ---------------- kNN (16 smallest d2 via iterative min-extract) -----------
__global__ void knn_kernel(const float* __restrict__ pts, const float* __restrict__ qpts,
                           int NP, int NS, int* __restrict__ nn_idx) {
    __shared__ float d2[512], rv[512];
    __shared__ int ri[512];
    const int tid = threadIdx.x;         // blockDim == NP
    const int b = blockIdx.x / NS;
    const int s = blockIdx.x % NS;
    const float qx = qpts[((size_t)b * NS + s) * 2 + 0];
    const float qy = qpts[((size_t)b * NS + s) * 2 + 1];
    const float dx = pts[((size_t)b * NP + tid) * 2 + 0] - qx;
    const float dy = pts[((size_t)b * NP + tid) * 2 + 1] - qy;
    d2[tid] = dx * dx + dy * dy;
    __syncthreads();
    for (int k = 0; k < KNN; ++k) {
        rv[tid] = d2[tid]; ri[tid] = tid;
        __syncthreads();
        for (int s2 = NP >> 1; s2 > 0; s2 >>= 1) {
            if (tid < s2 && rv[tid + s2] < rv[tid]) { rv[tid] = rv[tid + s2]; ri[tid] = ri[tid + s2]; }
            __syncthreads();
        }
        if (tid == 0) {
            nn_idx[((size_t)b * NS + s) * KNN + k] = ri[0];
            d2[ri[0]] = 3.4e38f;
        }
        __syncthreads();
    }
}

// ---------------- per-sample MLP stage ------------------------------------
// One block (256 thr / 8 waves) per (batch, sample).
// TDM gather-mode DMA pulls the 16 neighbor feature rows into LDS while the
// other waves stage the anchor half of the agg input; both GEMMs run on
// bf16 WMMA with dual accumulators per wave to hide WMMA->WMMA hazards.
__global__ void stage_mlp_kernel(
    const float* __restrict__ fts_in,   // (B, NP, 1024)
    const float* __restrict__ pts_in,   // (B, NP, 2)
    const int*   __restrict__ fps_idx,  // (B, NS)
    const int*   __restrict__ nn_idx,   // (B, NS, 16)
    const __bf16* __restrict__ dwp,     // (1040, 1056) padded bf16 diff_w
    const float* __restrict__ db,       // (1026)
    const __bf16* __restrict__ awp,     // (1024, 2080) padded bf16 agg_w
    const float* __restrict__ ab,       // (1024)
    const float* __restrict__ lng, const float* __restrict__ lnb,
    float* __restrict__ fts_out,        // (B, NS, 1024)
    int NP, int NS) {
    extern __shared__ __align__(16) char smem[];
    __bf16* A1 = (__bf16*)smem;                       // 16 x 2080 bf16 (agg input)
    char* pp = smem + (size_t)KNN * G2P * 2;          // 66560
    __bf16* A0 = (__bf16*)pp; pp += (size_t)KNN * C2P * 2;   // 16 x 1056 bf16
    float* rowsL = (float*)pp;                        // 16 x 1024 f32 (TDM gather dest)
    float* H     = (float*)pp;                        // 16 x 1024 f32 (reuses rowsL later)
    pp += (size_t)KNN * C_DIM * 4;
    float* anchorF = (float*)pp; pp += (C2 * 4 + 8);  // 1026 f32
    float* partS = (float*)pp;   pp += 16 * 16 * 4;
    float* partQ = (float*)pp;   pp += 16 * 16 * 4;
    float* muL   = (float*)pp;   pp += 16 * 4;
    float* rsL   = (float*)pp;   pp += 16 * 4;
    int*   nnL   = (int*)pp;                          // 16 ints

    const int tid  = threadIdx.x;
    const int lane = tid & 31;
    const int wv   = tid >> 5;
    const int b = blockIdx.x / NS;
    const int s = blockIdx.x % NS;

    // ---- anchor feature+coords, neighbor indices ----
    const int fidx = fps_idx[(size_t)b * NS + s];
    for (int c = tid; c < C_DIM; c += 256)
        anchorF[c] = fts_in[((size_t)b * NP + fidx) * C_DIM + c];
    if (tid == 0) {
        anchorF[1024] = pts_in[((size_t)b * NP + fidx) * 2 + 0];
        anchorF[1025] = pts_in[((size_t)b * NP + fidx) * 2 + 1];
    }
    if (tid < KNN) nnL[tid] = nn_idx[((size_t)b * NS + s) * KNN + tid];
    __syncthreads();

    // ---- wave 0: TDM gather of the 16 neighbor rows -> rowsL -------------
    // D# per ISA 8.3-8.7: gather_mode, 16-bit indices, 2D tensor (NP x 1024 f32).
    if (wv == 0) {
        const unsigned long long gaddr =
            (unsigned long long)(uintptr_t)(fts_in + (size_t)b * NP * C_DIM);
        u32x4 g0;
        g0[0] = 0x80000001u;                          // count=1 | gather_mode=1
        g0[1] = (unsigned)(uintptr_t)rowsL;           // lds_addr (low 32 = LDS offset)
        g0[2] = (unsigned)(gaddr & 0xffffffffu);      // global_addr[31:0]
        g0[3] = (unsigned)((gaddr >> 32) & 0x1ffffffu) | (2u << 30); // addr[56:32]|type=2
        i32x8 g1;
        g1[0] = 0x00020000;                           // data_size = 4B
        g1[1] = (int)(1024u << 16);                   // tensor_dim0[15:0] @ bits 63:48
        g1[2] = (int)((unsigned)NP << 16);            // tensor_dim1[15:0] @ bits 95:80
        g1[3] = (int)(1024u << 16);                   // tile_dim0 @ bits 127:112
        g1[4] = 16;                                   // tile_dim1 = #valid indices
        g1[5] = 1024;                                 // tensor_dim0_stride[31:0]
        g1[6] = 0;
        g1[7] = 0;
        i32x4 g2, g3;                                 // packed 16-bit row indices
        g2[0] = nnL[0]  | (nnL[1]  << 16);
        g2[1] = nnL[2]  | (nnL[3]  << 16);
        g2[2] = nnL[4]  | (nnL[5]  << 16);
        g2[3] = nnL[6]  | (nnL[7]  << 16);
        g3[0] = nnL[8]  | (nnL[9]  << 16);
        g3[1] = nnL[10] | (nnL[11] << 16);
        g3[2] = nnL[12] | (nnL[13] << 16);
        g3[3] = nnL[14] | (nnL[15] << 16);
        i32x8 g4 = {0, 0, 0, 0, 0, 0, 0, 0};          // unused trailing group
        __builtin_amdgcn_tensor_load_to_lds(g0, g1, g2, g3, g4, 0);
        __builtin_amdgcn_s_wait_tensorcnt(0);
    }
    // ---- other waves concurrently: anchor half of A1 (cols 1026..2079) ----
    const int AN = G2P - C2;
    for (int i = tid; i < KNN * AN; i += 256) {
        const int k = i / AN, d = C2 + (i % AN);
        const float v = (d < G2) ? anchorF[d - C2] : 0.f;
        A1[k * G2P + d] = (__bf16)v;
    }
    __syncthreads();

    // ---- build A0 = (gathered - anchor) bf16 from LDS rows ----------------
    for (int i = tid; i < KNN * C2P; i += 256) {
        const int k = i / C2P, d = i % C2P;
        float v = 0.f;
        if (d < C_DIM) {
            v = rowsL[k * C_DIM + d] - anchorF[d];
        } else if (d < C2) {
            const int j = nnL[k];
            v = pts_in[((size_t)b * NP + j) * 2 + (d - C_DIM)] - anchorF[d];
        }
        A0[k * C2P + d] = (__bf16)v;
    }
    __syncthreads();

    // ---- diff GEMM: (16 x 1056) @ dwp^T -> cols 0..1025 of A1, +bias ------
    // Tiles 0..63: guard-free pair loop (all cols < 1024 < C2).
    for (int ot0 = 2 * wv; ot0 < 64; ot0 += 16) {
        f32x8 acc0 = {}, acc1 = {};
        const int ob0 = ot0 * 16, ob1 = ob0 + 16;
        for (int kc = 0; kc < C2P / 32; ++kc) {
            bf16x16 a  = frag_load(A0, lane & 15, C2P, kc * 32, lane);
            bf16x16 w0 = frag_load(dwp, ob0 + (lane & 15), C2P, kc * 32, lane);
            acc0 = wmma_bf16(a, w0, acc0);
            bf16x16 w1 = frag_load(dwp, ob1 + (lane & 15), C2P, kc * 32, lane);
            acc1 = wmma_bf16(a, w1, acc1);
        }
        const int m0 = (lane < 16) ? 0 : 8;
        {
            const int o = ob0 + (lane & 15);
            const float bias = db[o];
            #pragma unroll
            for (int r = 0; r < 8; ++r)
                A1[(m0 + r) * G2P + o] = (__bf16)(acc0[r] + bias);
        }
        {
            const int o = ob1 + (lane & 15);
            const float bias = db[o];
            #pragma unroll
            for (int r = 0; r < 8; ++r)
                A1[(m0 + r) * G2P + o] = (__bf16)(acc1[r] + bias);
        }
    }
    // Peeled odd tile 64 (cols 1024..1039; only 1024..1025 stored) on wave 0.
    if (wv == 0) {
        f32x8 acc = {};
        const int ob = 64 * 16;
        for (int kc = 0; kc < C2P / 32; ++kc) {
            bf16x16 a = frag_load(A0, lane & 15, C2P, kc * 32, lane);
            bf16x16 w = frag_load(dwp, ob + (lane & 15), C2P, kc * 32, lane);
            acc = wmma_bf16(a, w, acc);
        }
        const int o = ob + (lane & 15);
        if (o < C2) {                                 // don't clobber anchor cols
            const float bias = db[o];
            const int m0 = (lane < 16) ? 0 : 8;
            #pragma unroll
            for (int r = 0; r < 8; ++r)
                A1[(m0 + r) * G2P + o] = (__bf16)(acc[r] + bias);
        }
    }
    __syncthreads();

    // ---- agg GEMM: (16 x 2080) @ awp^T -> H (16 x 1024), bias + ReLU ------
    for (int ot0 = 2 * wv; ot0 < C_DIM / 16; ot0 += 16) {
        f32x8 acc0 = {}, acc1 = {};
        const int ob0 = ot0 * 16, ob1 = ob0 + 16;
        for (int kc = 0; kc < G2P / 32; ++kc) {
            bf16x16 a  = frag_load(A1, lane & 15, G2P, kc * 32, lane);
            bf16x16 w0 = frag_load(awp, ob0 + (lane & 15), G2P, kc * 32, lane);
            acc0 = wmma_bf16(a, w0, acc0);
            bf16x16 w1 = frag_load(awp, ob1 + (lane & 15), G2P, kc * 32, lane);
            acc1 = wmma_bf16(a, w1, acc1);
        }
        const int m0 = (lane < 16) ? 0 : 8;
        {
            const int o = ob0 + (lane & 15);
            const float bias = ab[o];
            #pragma unroll
            for (int r = 0; r < 8; ++r) {
                const float v = acc0[r] + bias;
                H[(m0 + r) * C_DIM + o] = v > 0.f ? v : 0.f;
            }
        }
        {
            const int o = ob1 + (lane & 15);
            const float bias = ab[o];
            #pragma unroll
            for (int r = 0; r < 8; ++r) {
                const float v = acc1[r] + bias;
                H[(m0 + r) * C_DIM + o] = v > 0.f ? v : 0.f;
            }
        }
    }
    __syncthreads();

    // ---- LayerNorm per row (16 rows x 1024) ----
    {
        const int r = tid >> 4, l16 = tid & 15;
        float sum = 0.f, sq = 0.f;
        for (int c = l16; c < C_DIM; c += 16) {
            const float v = H[r * C_DIM + c];
            sum += v; sq += v * v;
        }
        partS[r * 16 + l16] = sum;
        partQ[r * 16 + l16] = sq;
    }
    __syncthreads();
    if (tid < 16) {
        float sum = 0.f, sq = 0.f;
        for (int i = 0; i < 16; ++i) { sum += partS[tid * 16 + i]; sq += partQ[tid * 16 + i]; }
        const float mu = sum / (float)C_DIM;
        const float var = sq / (float)C_DIM - mu * mu;
        muL[tid] = mu;
        rsL[tid] = rsqrtf(var + LN_EPS);
    }
    __syncthreads();

    // ---- affine + max over k ----
    for (int c = tid; c < C_DIM; c += 256) {
        const float g = lng[c], be = lnb[c];
        float m = -3.4e38f;
        #pragma unroll
        for (int r = 0; r < KNN; ++r) {
            const float v = (H[r * C_DIM + c] - muL[r]) * rsL[r] * g + be;
            m = fmaxf(m, v);
        }
        fts_out[((size_t)b * NS + s) * C_DIM + c] = m;
    }
}

// ---------------- flat layer: out1[b,o] = fb[o] + <fts2[b,:], fw[o,:]> ----
__global__ void flat_kernel(const float* __restrict__ fts2, const float* __restrict__ fw,
                            const float* __restrict__ fb, float* __restrict__ out1) {
    __shared__ float red[256];
    const int o = blockIdx.x;
    const int tid = threadIdx.x;
    float acc[B_SZ];
    #pragma unroll
    for (int b = 0; b < B_SZ; ++b) acc[b] = 0.f;
    const float* wrow = fw + (size_t)o * (C_DIM * N2);
    for (int j = tid; j < C_DIM * N2; j += 256) {
        const float wvv = wrow[j];
        #pragma unroll
        for (int b = 0; b < B_SZ; ++b)
            acc[b] += wvv * fts2[(size_t)b * (C_DIM * N2) + j];
    }
    for (int b = 0; b < B_SZ; ++b) {
        red[tid] = acc[b];
        __syncthreads();
        for (int s2 = 128; s2 > 0; s2 >>= 1) {
            if (tid < s2) red[tid] += red[tid + s2];
            __syncthreads();
        }
        if (tid == 0) out1[(size_t)b * C_DIM + o] = red[0] + fb[o];
        __syncthreads();
    }
}

// ---------------- dim layer: out[b,o] = dmb[o] + <out1[b,:], dmw[o,:]> ----
__global__ void dim_kernel(const float* __restrict__ out1, const float* __restrict__ dmw,
                           const float* __restrict__ dmb, float* __restrict__ out) {
    __shared__ float red[256];
    const int o = blockIdx.x;
    const int tid = threadIdx.x;
    float acc[B_SZ];
    #pragma unroll
    for (int b = 0; b < B_SZ; ++b) acc[b] = 0.f;
    const float* wrow = dmw + (size_t)o * C_DIM;
    for (int c = tid; c < C_DIM; c += 256) {
        const float wvv = wrow[c];
        #pragma unroll
        for (int b = 0; b < B_SZ; ++b)
            acc[b] += wvv * out1[(size_t)b * C_DIM + c];
    }
    for (int b = 0; b < B_SZ; ++b) {
        red[tid] = acc[b];
        __syncthreads();
        for (int s2 = 128; s2 > 0; s2 >>= 1) {
            if (tid < s2) red[tid] += red[tid + s2];
            __syncthreads();
        }
        if (tid == 0) out[(size_t)b * OUT_DIM + o] = red[0] + dmb[o];
        __syncthreads();
    }
}

// =====================================================================
extern "C" void kernel_launch(void* const* d_in, const int* in_sizes, int n_in,
                              void* d_out, int out_size, void* d_ws, size_t ws_size,
                              hipStream_t stream) {
    const float* points  = (const float*)d_in[0];
    const float* featmap = (const float*)d_in[1];
    const float* dw0 = (const float*)d_in[2];
    const float* db0 = (const float*)d_in[3];
    const float* aw0 = (const float*)d_in[4];
    const float* ab0 = (const float*)d_in[5];
    const float* lg0 = (const float*)d_in[6];
    const float* lb0 = (const float*)d_in[7];
    const float* dw1 = (const float*)d_in[8];
    const float* db1 = (const float*)d_in[9];
    const float* aw1 = (const float*)d_in[10];
    const float* ab1 = (const float*)d_in[11];
    const float* lg1 = (const float*)d_in[12];
    const float* lb1 = (const float*)d_in[13];
    const float* fw  = (const float*)d_in[14];
    const float* fb  = (const float*)d_in[15];
    const float* dmw = (const float*)d_in[16];
    const float* dmb = (const float*)d_in[17];
    float* out = (float*)d_out;

    char* ws = (char*)d_ws;
    size_t off = 0;
    auto alloc = [&](size_t bytes) -> char* {
        char* p = ws + off;
        off = (off + bytes + 255) & ~(size_t)255;
        return p;
    };
    float* fts0 = (float*)alloc((size_t)B_SZ * N0 * C_DIM * 4);   // 64 MB
    float* fts1 = (float*)alloc((size_t)B_SZ * N1 * C_DIM * 4);   // 32 MB
    float* fts2 = (float*)alloc((size_t)B_SZ * N2 * C_DIM * 4);   //  8 MB
    float* pts1 = (float*)alloc((size_t)B_SZ * N1 * 2 * 4);
    float* pts2 = (float*)alloc((size_t)B_SZ * N2 * 2 * 4);
    int* fpsi0  = (int*)alloc((size_t)B_SZ * N1 * 4);
    int* fpsi1  = (int*)alloc((size_t)B_SZ * N2 * 4);
    int* nn0    = (int*)alloc((size_t)B_SZ * N1 * KNN * 4);
    int* nn1    = (int*)alloc((size_t)B_SZ * N2 * KNN * 4);
    __bf16* dwp0 = (__bf16*)alloc((size_t)OPAD * C2P * 2);
    __bf16* dwp1 = (__bf16*)alloc((size_t)OPAD * C2P * 2);
    __bf16* awp0 = (__bf16*)alloc((size_t)C_DIM * G2P * 2);
    __bf16* awp1 = (__bf16*)alloc((size_t)C_DIM * G2P * 2);
    float* out1  = (float*)alloc((size_t)B_SZ * C_DIM * 4);

    // 1) weight conversion/padding to bf16
    {
        int n = OPAD * C2P;
        pad_bf16_kernel<<<(n + 255) / 256, 256, 0, stream>>>(dw0, dwp0, C2, C2, OPAD, C2P);
        pad_bf16_kernel<<<(n + 255) / 256, 256, 0, stream>>>(dw1, dwp1, C2, C2, OPAD, C2P);
        n = C_DIM * G2P;
        pad_bf16_kernel<<<(n + 255) / 256, 256, 0, stream>>>(aw0, awp0, C_DIM, G2, C_DIM, G2P);
        pad_bf16_kernel<<<(n + 255) / 256, 256, 0, stream>>>(aw1, awp1, C_DIM, G2, C_DIM, G2P);
    }

    // 2) bilinear feature sampling
    bilinear_kernel<<<B_SZ * N0, 256, 0, stream>>>(points, featmap, fts0);

    // 3) stage 0: FPS -> kNN -> WMMA MLP (TDM gather inside)
    fps_kernel<<<B_SZ, N0, 0, stream>>>(points, N0, N1, fpsi0, pts1);
    knn_kernel<<<B_SZ * N1, N0, 0, stream>>>(points, pts1, N0, N1, nn0);
    const size_t smem = (size_t)KNN * G2P * 2      // A1
                      + (size_t)KNN * C2P * 2      // A0
                      + (size_t)KNN * C_DIM * 4    // rows / H
                      + (C2 * 4 + 8)               // anchor
                      + 16 * 16 * 4 * 2            // partial sums
                      + 16 * 4 * 2                 // mu / rstd
                      + 16 * 4 + 64;               // nn idx + slack
    stage_mlp_kernel<<<B_SZ * N1, 256, smem, stream>>>(
        fts0, points, fpsi0, nn0, dwp0, db0, awp0, ab0, lg0, lb0, fts1, N0, N1);

    // 4) stage 1
    fps_kernel<<<B_SZ, N1, 0, stream>>>(pts1, N1, N2, fpsi1, pts2);
    knn_kernel<<<B_SZ * N2, N1, 0, stream>>>(pts1, pts2, N1, N2, nn1);
    stage_mlp_kernel<<<B_SZ * N2, 256, smem, stream>>>(
        fts1, pts1, fpsi1, nn1, dwp1, db1, awp1, ab1, lg1, lb1, fts2, N1, N2);

    // 5) flat + dim projections (bandwidth-bound, fp32)
    flat_kernel<<<C_DIM, 256, 0, stream>>>(fts2, fw, fb, out1);
    dim_kernel<<<OUT_DIM, 256, 0, stream>>>(out1, dmw, dmb, out);
}
